// LSTM2Layer_64424509440171
// MI455X (gfx1250) — matrix operand order repaired and checked by reading, not compiled, not run on hardware
//
#include <hip/hip_runtime.h>

typedef __attribute__((ext_vector_type(16))) _Float16 v16h;
typedef __attribute__((ext_vector_type(8)))  _Float16 v8h;
typedef __attribute__((ext_vector_type(8)))  float    v8f;

#define B_  4096
#define T_  256
#define D_  16
#define H_  32

#define LOG2E_F  1.44269504088896f

// Bias-folded fast activations (TRANS-unit exp/rcp, fma folds the bias in free).
// sb = -bias*LOG2E ; tb = 2*bias*LOG2E
__device__ __forceinline__ float sigmoid_b(float x, float sb) {
    return __builtin_amdgcn_rcpf(1.0f + __builtin_amdgcn_exp2f(__builtin_fmaf(x, -LOG2E_F, sb)));
}
__device__ __forceinline__ float tanh_b(float x, float tb) {
    float r = __builtin_amdgcn_rcpf(1.0f + __builtin_amdgcn_exp2f(__builtin_fmaf(x, 2.0f * LOG2E_F, tb)));
    return 1.0f - 2.0f * r;
}
__device__ __forceinline__ float tanh_(float x) {
    float r = __builtin_amdgcn_rcpf(1.0f + __builtin_amdgcn_exp2f(x * (2.0f * LOG2E_F)));
    return 1.0f - 2.0f * r;
}

__device__ __forceinline__ void cvt4(v16h& d, int base, float4 a) {
    d[base + 0] = (_Float16)a.x; d[base + 1] = (_Float16)a.y;
    d[base + 2] = (_Float16)a.z; d[base + 3] = (_Float16)a.w;
}

__global__ __launch_bounds__(32, 1)
void lstm2_fused_wmma(const float* __restrict__ x,
                      const float* __restrict__ w_ih1, const float* __restrict__ w_hh1,
                      const float* __restrict__ b_ih1, const float* __restrict__ b_hh1,
                      const float* __restrict__ w_ih2, const float* __restrict__ w_hh2,
                      const float* __restrict__ b_ih2, const float* __restrict__ b_hh2,
                      const float* __restrict__ fc_w, const float* __restrict__ fc_b,
                      float* __restrict__ out)
{
    __shared__ __align__(16) _Float16 hbuf1[16 * 32];
    __shared__ __align__(16) _Float16 hbuf2[16 * 32];

    const int lane  = threadIdx.x;   // 0..31 (wave32)
    const int lo    = lane & 15;
    const int hi    = lane >> 4;
    const int bbase = blockIdx.x * 16;
    const int M     = lo;

    // ---------------- Resident B-layout weight registers ----------------
    v16h wih1[8], whh1[8], wih2[8], whh2[8];
    // pre-scaled per-tile biases: sigmoid tiles (i,f,o = 0,1,2,3,6,7) use sb,
    // tanh tiles (g = 4,5) use tb.
    float sb1[8], tb1[8], sb2[8], tb2[8];
    const int kb = hi * 16;
    #pragma unroll
    for (int n = 0; n < 8; ++n) {
        const int g = n * 16 + lo;
        const float4* ph1 = (const float4*)(w_hh1 + g * H_ + kb);
        const float4* ph2 = (const float4*)(w_hh2 + g * H_ + kb);
        const float4* pi2 = (const float4*)(w_ih2 + g * H_ + kb);
        const float4* pi1 = (const float4*)(w_ih1 + g * D_ + hi * 8);
        #pragma unroll
        for (int q = 0; q < 4; ++q) {
            cvt4(whh1[n], 4 * q, ph1[q]);
            cvt4(whh2[n], 4 * q, ph2[q]);
            cvt4(wih2[n], 4 * q, pi2[q]);
        }
        float4 i1a = pi1[0], i1b = pi1[1], i1c = make_float4(0.f, 0.f, 0.f, 0.f), i1d = i1c;
        cvt4(wih1[n], 0, i1a); cvt4(wih1[n], 4, i1b);
        cvt4(wih1[n], 8, i1c); cvt4(wih1[n], 12, i1d);
        float b1 = b_ih1[g] + b_hh1[g];
        float b2 = b_ih2[g] + b_hh2[g];
        sb1[n] = b1 * -LOG2E_F;  tb1[n] = b1 * (2.0f * LOG2E_F);
        sb2[n] = b2 * -LOG2E_F;  tb2[n] = b2 * (2.0f * LOG2E_F);
    }

    // ---------------- State ----------------
    v16h h1a = {};                 // h1[t]   in A-layout f16
    v16h h2a = {};                 // h2[t-1] in A-layout f16
    v8f  c1[2] = {}, c2[2] = {};
    v8f  h2f[2] = {};
    const v8f zero8 = {};

    const float* xrow = x + (size_t)(bbase + M) * (T_ * D_);
    const int k0 = hi * 16;

    // ---------------- Prologue: layer1 step 0 (h1=c1=0) ----------------
    float4 x0, x1;
    {
        const float4* xp = (const float4*)(xrow + hi * 8);
        x0 = xp[0]; x1 = xp[1];
    }
    {
        v16h xa;
        cvt4(xa, 0, x0); cvt4(xa, 4, x1);
        #pragma unroll
        for (int j = 8; j < 16; ++j) xa[j] = (_Float16)0.0f;
        // prefetch x[1]
        const float4* xp = (const float4*)(xrow + 1 * D_ + hi * 8);
        x0 = xp[0]; x1 = xp[1];

        v8f acc1[8];
        #pragma unroll
        for (int n = 0; n < 8; ++n)
            acc1[n] = __builtin_amdgcn_wmma_f32_16x16x32_f16(false, xa, false, wih1[n], (short)0, zero8, false, false);
        #pragma unroll
        for (int n = 0; n < 2; ++n) {
            const int col = n * 16 + lo;
            #pragma unroll
            for (int e = 0; e < 8; ++e) {
                float iv = sigmoid_b(acc1[0 + n][e], sb1[0 + n]);
                float fv = sigmoid_b(acc1[2 + n][e], sb1[2 + n]);
                float gv = tanh_b   (acc1[4 + n][e], tb1[4 + n]);
                float ov = sigmoid_b(acc1[6 + n][e], sb1[6 + n]);
                float cv = fv * c1[n][e] + iv * gv;
                c1[n][e] = cv;
                hbuf1[(hi * 8 + e) * 32 + col] = (_Float16)(ov * tanh_(cv));
            }
        }
        v8h l8 = *(const v8h*)(hbuf1 + M * 32 + k0);
        v8h h8 = *(const v8h*)(hbuf1 + M * 32 + k0 + 8);
        #pragma unroll
        for (int j = 0; j < 8; ++j) { h1a[j] = l8[j]; h1a[8 + j] = h8[j]; }
    }

    // ---------------- Main loop: layer1 step t+1 || layer2 step t ----------------
    for (int t = 0; t < T_ - 1; ++t) {
        // xa for layer1 step t+1 (prefetched); prefetch x[t+2] (wraps, in-bounds)
        v16h xa;
        cvt4(xa, 0, x0); cvt4(xa, 4, x1);
        #pragma unroll
        for (int j = 8; j < 16; ++j) xa[j] = (_Float16)0.0f;
        {
            const int tn = (t + 2) & (T_ - 1);
            const float4* xp = (const float4*)(xrow + tn * D_ + hi * 8);
            x0 = xp[0]; x1 = xp[1];
        }

        // 32 WMMAs as 16 independent 2-deep chains: both recurrent chains advance together.
        v8f acc1[8], acc2[8];
        #pragma unroll
        for (int n = 0; n < 8; ++n) {
            v8f a = __builtin_amdgcn_wmma_f32_16x16x32_f16(false, xa,  false, wih1[n], (short)0, zero8, false, false);
            acc1[n] = __builtin_amdgcn_wmma_f32_16x16x32_f16(false, h1a, false, whh1[n], (short)0, a, false, false);
        }
        #pragma unroll
        for (int n = 0; n < 8; ++n) {
            v8f a = __builtin_amdgcn_wmma_f32_16x16x32_f16(false, h1a, false, wih2[n], (short)0, zero8, false, false);
            acc2[n] = __builtin_amdgcn_wmma_f32_16x16x32_f16(false, h2a, false, whh2[n], (short)0, a, false, false);
        }

        // ---- layer1 elementwise (step t+1) -> c1, hbuf1 -> h1a ----
        #pragma unroll
        for (int n = 0; n < 2; ++n) {
            const int col = n * 16 + lo;
            #pragma unroll
            for (int e = 0; e < 8; ++e) {
                float iv = sigmoid_b(acc1[0 + n][e], sb1[0 + n]);
                float fv = sigmoid_b(acc1[2 + n][e], sb1[2 + n]);
                float gv = tanh_b   (acc1[4 + n][e], tb1[4 + n]);
                float ov = sigmoid_b(acc1[6 + n][e], sb1[6 + n]);
                float cv = fv * c1[n][e] + iv * gv;
                c1[n][e] = cv;
                hbuf1[(hi * 8 + e) * 32 + col] = (_Float16)(ov * tanh_(cv));
            }
        }
        {
            v8h l8 = *(const v8h*)(hbuf1 + M * 32 + k0);
            v8h h8 = *(const v8h*)(hbuf1 + M * 32 + k0 + 8);
            #pragma unroll
            for (int j = 0; j < 8; ++j) { h1a[j] = l8[j]; h1a[8 + j] = h8[j]; }
        }

        // ---- layer2 elementwise (step t) -> c2, h2f, hbuf2 -> h2a ----
        #pragma unroll
        for (int n = 0; n < 2; ++n) {
            const int col = n * 16 + lo;
            #pragma unroll
            for (int e = 0; e < 8; ++e) {
                float iv = sigmoid_b(acc2[0 + n][e], sb2[0 + n]);
                float fv = sigmoid_b(acc2[2 + n][e], sb2[2 + n]);
                float gv = tanh_b   (acc2[4 + n][e], tb2[4 + n]);
                float ov = sigmoid_b(acc2[6 + n][e], sb2[6 + n]);
                float cv = fv * c2[n][e] + iv * gv;
                c2[n][e] = cv;
                float hv = ov * tanh_(cv);
                h2f[n][e] = hv;
                hbuf2[(hi * 8 + e) * 32 + col] = (_Float16)hv;
            }
        }
        {
            v8h l8 = *(const v8h*)(hbuf2 + M * 32 + k0);
            v8h h8 = *(const v8h*)(hbuf2 + M * 32 + k0 + 8);
            #pragma unroll
            for (int j = 0; j < 8; ++j) { h2a[j] = l8[j]; h2a[8 + j] = h8[j]; }
        }
    }

    // ---------------- Epilogue: layer2 step T-1 (no LDS round-trip needed) ----------------
    {
        v8f acc2[8];
        #pragma unroll
        for (int n = 0; n < 8; ++n) {
            v8f a = __builtin_amdgcn_wmma_f32_16x16x32_f16(false, h1a, false, wih2[n], (short)0, zero8, false, false);
            acc2[n] = __builtin_amdgcn_wmma_f32_16x16x32_f16(false, h2a, false, whh2[n], (short)0, a, false, false);
        }
        #pragma unroll
        for (int n = 0; n < 2; ++n) {
            #pragma unroll
            for (int e = 0; e < 8; ++e) {
                float iv = sigmoid_b(acc2[0 + n][e], sb2[0 + n]);
                float fv = sigmoid_b(acc2[2 + n][e], sb2[2 + n]);
                float gv = tanh_b   (acc2[4 + n][e], tb2[4 + n]);
                float ov = sigmoid_b(acc2[6 + n][e], sb2[6 + n]);
                float cv = fv * c2[n][e] + iv * gv;
                h2f[n][e] = ov * tanh_(cv);
            }
        }
    }

    // ---------------- FC head: out[b] = h2[b,:] . fc_w + fc_b ----------------
    const float fw0 = fc_w[lo];
    const float fw1 = fc_w[16 + lo];
    const float fb  = fc_b[0];
    float p[8];
    #pragma unroll
    for (int e = 0; e < 8; ++e) {
        float s = h2f[0][e] * fw0 + h2f[1][e] * fw1;
        s += __shfl_xor(s, 1, 32);
        s += __shfl_xor(s, 2, 32);
        s += __shfl_xor(s, 4, 32);
        s += __shfl_xor(s, 8, 32);
        p[e] = s;
    }
    if (lo == 0) {
        #pragma unroll
        for (int e = 0; e < 8; ++e)
            out[bbase + hi * 8 + e] = p[e] + fb;
    }
}

extern "C" void kernel_launch(void* const* d_in, const int* in_sizes, int n_in,
                              void* d_out, int out_size, void* d_ws, size_t ws_size,
                              hipStream_t stream) {
    (void)in_sizes; (void)n_in; (void)out_size; (void)d_ws; (void)ws_size;
    const float* x     = (const float*)d_in[0];
    const float* w_ih1 = (const float*)d_in[1];
    const float* w_hh1 = (const float*)d_in[2];
    const float* b_ih1 = (const float*)d_in[3];
    const float* b_hh1 = (const float*)d_in[4];
    const float* w_ih2 = (const float*)d_in[5];
    const float* w_hh2 = (const float*)d_in[6];
    const float* b_ih2 = (const float*)d_in[7];
    const float* b_hh2 = (const float*)d_in[8];
    const float* fc_w  = (const float*)d_in[9];
    const float* fc_b  = (const float*)d_in[10];
    float* out = (float*)d_out;

    dim3 grid(B_ / 16), block(32);
    lstm2_fused_wmma<<<grid, block, 0, stream>>>(
        x, w_ih1, w_hh1, b_ih1, b_hh1,
        w_ih2, w_hh2, b_ih2, b_hh2, fc_w, fc_b, out);
}
